// WordSentAtt_80410377715991
// MI455X (gfx1250) — matrix operand order repaired and checked by reading, not compiled, not run on hardware
//
#include <hip/hip_runtime.h>

// ---------------------------------------------------------------------------
// WordSentAtt for MI455X (gfx1250, wave32, WMMA bf16, async + TDM staging)
//   prep:   Qb = bf16(Q); Wb = bf16(W); Kb = bf16(key * mask)
//   gemm1:  Qf = bf16(relu(Qb @ Wb^T + b))     [WMMA bf16, async->LDS]
//   gemm2:  S  = Qf @ Kb^T + (mask-1)*NEG_BIG  [WMMA bf16, TDM tensor_load_to_lds]
//   smax:   P  = bf16(softmax_row(S))
//   gemm3:  O  = P @ Kb                        [WMMA bf16, async->LDS + LDS transpose]
// All GEMMs: 128x128 workgroup tile, 8 waves x (16x128), K-chunk 32,
// double-buffered LDS (80B row pitch -> ds_load_b128 fragment reads, no bank
// conflicts). TDM descriptor uses pad_enable to generate the same pitch.
// ---------------------------------------------------------------------------

#define NEG_BIG_F 10000000000000.0f

static constexpr int BB  = 32;    // batch
static constexpr int LQ  = 1024;  // query len
static constexpr int LKK = 1024;  // key len
static constexpr int DD  = 1024;  // feature dim
static constexpr int KP  = 40;    // LDS row pitch in u16 (32 data + 8 pad = 80B)

typedef __attribute__((ext_vector_type(16))) __bf16 v16bf;
typedef __attribute__((ext_vector_type(8)))  float  v8f;
typedef __attribute__((ext_vector_type(4)))  unsigned u32x4;
typedef __attribute__((ext_vector_type(8)))  unsigned u32x8;

union FragAB { v16bf v; unsigned u[8]; };

static __device__ __forceinline__ unsigned short f32_bf16(float f) {
  union { float f; unsigned u; } v; v.f = f;
  unsigned r = v.u + 0x7FFFu + ((v.u >> 16) & 1u);   // round-to-nearest-even
  return (unsigned short)(r >> 16);
}

// A fragment: 16x32 bf16. Lanes 0-15 rows M=0..15, K pairs {0..7,16..23};
// lanes 16-31 same rows, K pairs {8..15,24..31}. -> two ds_load_b128 per frag.
static __device__ __forceinline__ v16bf load_a_frag(const unsigned short* tile, int lane) {
  FragAB f;
  const int row = lane & 15;
  const int hi  = lane >> 4;
#pragma unroll
  for (int j = 0; j < 8; ++j) {
    const int k0 = ((j >> 2) << 4) + (hi << 3) + ((j & 3) << 1);
    f.u[j] = *(const unsigned*)(tile + row * KP + k0);
  }
  return f.v;
}

// B fragment: 32x16 bf16. Same K order as A: lanes 0-15 K {0..7,16..23},
// lanes 16-31 K {8..15,24..31} (N=lane&15). Tile stored [n][k], k-contiguous.
static __device__ __forceinline__ v16bf load_b_frag(const unsigned short* tile, int lane) {
  FragAB f;
  const int n  = lane & 15;
  const int hi = lane >> 4;
#pragma unroll
  for (int j = 0; j < 8; ++j) {
    const int k0 = ((j >> 2) << 4) + (hi << 3) + ((j & 3) << 1);
    f.u[j] = *(const unsigned*)(tile + n * KP + k0);
  }
  return f.v;
}

// ---- async global->LDS staging (ASYNCcnt) --------------------------------
static __device__ __forceinline__ void stage_async(const unsigned short* __restrict__ g,
                                                   long row0, long ldg, int kt,
                                                   unsigned lds_base, int tid) {
#pragma unroll
  for (int t = 0; t < 2; ++t) {
    const int id = tid + t * 256;
    const int r  = id >> 2;
    const int c8 = (id & 3) << 3;                          // 0,8,16,24
    const unsigned lds_off = lds_base + (unsigned)((r * KP + c8) * 2);
    const unsigned long long ga =
        (unsigned long long)(g + (row0 + r) * ldg + kt + c8);
    asm volatile("global_load_async_to_lds_b128 %0, %1, off"
                 :: "v"(lds_off), "v"(ga) : "memory");
  }
}

static __device__ __forceinline__ void wait_async0() {
  asm volatile("s_wait_asynccnt 0" ::: "memory");
}

// ---- TDM staging (TENSORcnt) ---------------------------------------------
// One tensor_load_to_lds DMAs a whole 128x32 bf16 tile. D# group0/group1 per
// CDNA5 ISA. pad: +16B after every 64B stored
// => LDS row pitch = 80B = KP u16, matching the fragment-read layout.
static __device__ __forceinline__ void tdm_load_tile(const unsigned short* gbase,
                                                     unsigned lds_off,
                                                     unsigned tensor_rows) {
  const unsigned long long ga = (unsigned long long)gbase;
  u32x4 g0;
  g0[0] = 1u;                                     // count=1, user mode
  g0[1] = lds_off;                                // lds_addr (bytes)
  g0[2] = (unsigned)ga;                           // global_addr[31:0]
  g0[3] = ((unsigned)(ga >> 32) & 0x01FFFFFFu)    // global_addr[56:32]
          | (2u << 30);                           // type=2 ("image")
  u32x8 g1;
  g1[0] = (1u << 16)                              // data_size=1 -> 2 bytes
        | (1u << 20)                              // pad_enable
        | (3u << 22)                              // pad_interval: 2^(3+1)=16 DW = 64B
        | (3u << 25);                             // pad_amount: 3+1 = 4 DW = 16B
  g1[1] = (1024u & 0xFFFFu) << 16;                // tensor_dim0 low16 (row len 1024)
  g1[2] = (1024u >> 16) | ((tensor_rows & 0xFFFFu) << 16);   // dim0 hi | dim1 lo
  g1[3] = ((tensor_rows >> 16) & 0xFFFFu) | (32u << 16);     // dim1 hi | tile_dim0=32
  g1[4] = 128u;                                   // tile_dim1=128, tile_dim2=0
  g1[5] = 1024u;                                  // tensor_dim0_stride low32
  g1[6] = 0u;                                     // stride hi | dim1_stride lo
  g1[7] = 0u;
  asm volatile("tensor_load_to_lds %0, %1" :: "s"(g0), "s"(g1) : "memory");
}

// One K-chunk of compute: preload A + all 8 B fragments, then issue the 8
// WMMAs back-to-back so the XDL pipe stays full.
static __device__ __forceinline__ void mma_chunk(const unsigned short* As,
                                                 const unsigned short* Bs,
                                                 int wave, int lane, v8f acc[8]) {
  const v16bf a = load_a_frag(As + (wave * 16) * KP, lane);
  v16bf bf[8];
#pragma unroll
  for (int i = 0; i < 8; ++i) bf[i] = load_b_frag(Bs + (i * 16) * KP, lane);
#pragma unroll
  for (int i = 0; i < 8; ++i)
    acc[i] = __builtin_amdgcn_wmma_f32_16x16x32_bf16(false, a, false, bf[i],
                                                     (short)0, acc[i], false, false);
}

// ---------------------------------------------------------------------------
// prep kernels (f32 -> bf16 conversions; memory-bound, ~17.5us total at 23.3TB/s)
// ---------------------------------------------------------------------------
__global__ __launch_bounds__(256) void k_prep_key(const float* __restrict__ key,
                                                  const float* __restrict__ mask,
                                                  unsigned short* __restrict__ Kb) {
  const long gid = ((long)blockIdx.x * 256 + threadIdx.x) * 4;
  const float mv = mask[gid >> 10];                  // row = b*LK + k (D=1024)
  const float4 kv = *(const float4*)(key + gid);
  uint2 pk;
  pk.x = (unsigned)f32_bf16(kv.x * mv) | ((unsigned)f32_bf16(kv.y * mv) << 16);
  pk.y = (unsigned)f32_bf16(kv.z * mv) | ((unsigned)f32_bf16(kv.w * mv) << 16);
  *(uint2*)(Kb + gid) = pk;
}

__global__ __launch_bounds__(256) void k_cvt(const float* __restrict__ src,
                                             unsigned short* __restrict__ dst) {
  const long gid = ((long)blockIdx.x * 256 + threadIdx.x) * 4;
  const float4 v = *(const float4*)(src + gid);
  uint2 pk;
  pk.x = (unsigned)f32_bf16(v.x) | ((unsigned)f32_bf16(v.y) << 16);
  pk.y = (unsigned)f32_bf16(v.z) | ((unsigned)f32_bf16(v.w) << 16);
  *(uint2*)(dst + gid) = pk;
}

// ---------------------------------------------------------------------------
// gemm1: Qf[m,n] = bf16(relu( sum_d Qb[m,d]*Wb[n,d] + bias[n] ))  [async]
// ---------------------------------------------------------------------------
__global__ __launch_bounds__(256) void k_gemm1(const unsigned short* __restrict__ Qb,
                                               const unsigned short* __restrict__ Wb,
                                               const float* __restrict__ bias,
                                               unsigned short* __restrict__ Qf) {
  __shared__ __align__(16) unsigned short As[2][128 * KP];
  __shared__ __align__(16) unsigned short Bs[2][128 * KP];
  const int tid = threadIdx.x, lane = tid & 31, wave = tid >> 5;
  const int n0 = blockIdx.x * 128;
  const long m0 = (long)blockIdx.y * 128;
  const unsigned aoff[2] = { (unsigned)(uintptr_t)&As[0][0], (unsigned)(uintptr_t)&As[1][0] };
  const unsigned boff[2] = { (unsigned)(uintptr_t)&Bs[0][0], (unsigned)(uintptr_t)&Bs[1][0] };

  v8f zero = {};
  v8f acc[8];
#pragma unroll
  for (int i = 0; i < 8; ++i) acc[i] = zero;

  stage_async(Qb, m0, DD, 0, aoff[0], tid);
  stage_async(Wb, n0, DD, 0, boff[0], tid);
  wait_async0();
  __syncthreads();

  for (int kt = 0; kt < DD; kt += 32) {
    const int cur = (kt >> 5) & 1, nxt = cur ^ 1;
    if (kt + 32 < DD) {
      stage_async(Qb, m0, DD, kt + 32, aoff[nxt], tid);
      stage_async(Wb, n0, DD, kt + 32, boff[nxt], tid);
    }
    mma_chunk(As[cur], Bs[cur], wave, lane, acc);
    wait_async0();
    __syncthreads();
  }

  const int mloc = (lane >> 4) << 3;
  const int nl = lane & 15;
#pragma unroll
  for (int i = 0; i < 8; ++i) {
    const int ng = n0 + i * 16 + nl;
    const float bv = bias[ng];
#pragma unroll
    for (int r = 0; r < 8; ++r) {
      const long mg = m0 + wave * 16 + mloc + r;
      float v = acc[i][r] + bv;
      v = v > 0.0f ? v : 0.0f;
      Qf[mg * DD + ng] = f32_bf16(v);
    }
  }
}

// ---------------------------------------------------------------------------
// gemm2: S[bq,n] = sum_d Qf[bq,d]*Kb[b,n,d] + (mask[b,n]-1)*NEG_BIG   [TDM]
//   wave0 DMAs the A tile, wave1 the B tile (TDM ignores EXEC; one wave each).
// ---------------------------------------------------------------------------
__global__ __launch_bounds__(256) void k_gemm2(const unsigned short* __restrict__ Qf,
                                               const unsigned short* __restrict__ Kb,
                                               const float* __restrict__ mask,
                                               float* __restrict__ S) {
  __shared__ __align__(16) unsigned short As[2][128 * KP];
  __shared__ __align__(16) unsigned short Bs[2][128 * KP];
  const int tid = threadIdx.x, lane = tid & 31, wave = tid >> 5;
  const int b = blockIdx.z;
  const int n0 = blockIdx.x * 128;
  const long m0 = (long)b * LQ + (long)blockIdx.y * 128;
  const unsigned short* KbB = Kb + ((long)b << 20);       // b*LK*DD
  const unsigned aoff[2] = { (unsigned)(uintptr_t)&As[0][0], (unsigned)(uintptr_t)&As[1][0] };
  const unsigned boff[2] = { (unsigned)(uintptr_t)&Bs[0][0], (unsigned)(uintptr_t)&Bs[1][0] };

  v8f zero = {};
  v8f acc[8];
#pragma unroll
  for (int i = 0; i < 8; ++i) acc[i] = zero;

  if (wave == 0) tdm_load_tile(Qf + m0 * DD, aoff[0], (unsigned)(BB * LQ));
  if (wave == 1) tdm_load_tile(KbB + (long)n0 * DD, boff[0], (unsigned)LKK);
  if (wave < 2) __builtin_amdgcn_s_wait_tensorcnt(0);
  __syncthreads();

  for (int kt = 0; kt < DD; kt += 32) {
    const int cur = (kt >> 5) & 1, nxt = cur ^ 1;
    if (kt + 32 < DD) {
      if (wave == 0) tdm_load_tile(Qf + m0 * DD + (kt + 32), aoff[nxt], (unsigned)(BB * LQ));
      if (wave == 1) tdm_load_tile(KbB + (long)n0 * DD + (kt + 32), boff[nxt], (unsigned)LKK);
    }
    mma_chunk(As[cur], Bs[cur], wave, lane, acc);
    if (wave < 2) __builtin_amdgcn_s_wait_tensorcnt(0);
    __syncthreads();
  }

  const int mloc = (lane >> 4) << 3;
  const int nl = lane & 15;
#pragma unroll
  for (int i = 0; i < 8; ++i) {
    const int ng = n0 + i * 16 + nl;                 // key position
    const float pen = (mask[(long)b * LKK + ng] - 1.0f) * NEG_BIG_F;
#pragma unroll
    for (int r = 0; r < 8; ++r) {
      const long mg = m0 + wave * 16 + mloc + r;
      S[mg * (long)LKK + ng] = acc[i][r] + pen;
    }
  }
}

// ---------------------------------------------------------------------------
// softmax: one wave per 1024-wide row; P = bf16(softmax(S_row))
// ---------------------------------------------------------------------------
__global__ __launch_bounds__(256) void k_softmax(const float* __restrict__ S,
                                                 unsigned short* __restrict__ P) {
  const int wave = threadIdx.x >> 5, lane = threadIdx.x & 31;
  const long row = (long)blockIdx.x * 8 + wave;
  const float* s = S + row * LKK;
  float x[32];
  float m = -3.4e38f;
#pragma unroll
  for (int i = 0; i < 32; ++i) { x[i] = s[i * 32 + lane]; m = fmaxf(m, x[i]); }
  for (int off = 16; off; off >>= 1) m = fmaxf(m, __shfl_xor(m, off, 32));
  float sum = 0.0f;
#pragma unroll
  for (int i = 0; i < 32; ++i) { x[i] = __expf(x[i] - m); sum += x[i]; }
  for (int off = 16; off; off >>= 1) sum += __shfl_xor(sum, off, 32);
  const float inv = 1.0f / sum;
  unsigned short* p = P + row * LKK;
#pragma unroll
  for (int i = 0; i < 32; ++i) p[i * 32 + lane] = f32_bf16(x[i] * inv);
}

// ---------------------------------------------------------------------------
// gemm3: O[bq,d] = sum_k P[bq,k]*Kb[b,k,d]
//   A staged async; B is k-major in memory -> manual transpose into LDS.
// ---------------------------------------------------------------------------
__global__ __launch_bounds__(256) void k_gemm3(const unsigned short* __restrict__ P,
                                               const unsigned short* __restrict__ Kb,
                                               float* __restrict__ O) {
  __shared__ __align__(16) unsigned short As[2][128 * KP];
  __shared__ __align__(16) unsigned short Bs[2][128 * KP];
  const int tid = threadIdx.x, lane = tid & 31, wave = tid >> 5;
  const int b = blockIdx.z;
  const int n0 = blockIdx.x * 128;                   // d tile
  const long m0 = (long)b * LQ + (long)blockIdx.y * 128;
  const unsigned short* KbB = Kb + ((long)b << 20);
  const unsigned aoff[2] = { (unsigned)(uintptr_t)&As[0][0], (unsigned)(uintptr_t)&As[1][0] };

  v8f zero = {};
  v8f acc[8];
#pragma unroll
  for (int i = 0; i < 8; ++i) acc[i] = zero;

  // B transpose staging: k-major rows -> LDS [n][k]
  auto stage_b = [&](unsigned short* Bt, int kt) {
#pragma unroll
    for (int t = 0; t < 8; ++t) {
      const int id = tid + t * 256;                  // 2048 u32 chunks
      const int kc = id >> 6;                        // 0..31
      const int n2 = (id & 63) << 1;                 // 0..126
      const unsigned v = *(const unsigned*)(KbB + (long)(kt + kc) * DD + n0 + n2);
      Bt[(n2 + 0) * KP + kc] = (unsigned short)(v & 0xFFFFu);
      Bt[(n2 + 1) * KP + kc] = (unsigned short)(v >> 16);
    }
  };

  stage_async(P, m0, LKK, 0, aoff[0], tid);
  stage_b(Bs[0], 0);
  wait_async0();
  __syncthreads();

  for (int kt = 0; kt < LKK; kt += 32) {
    const int cur = (kt >> 5) & 1, nxt = cur ^ 1;
    if (kt + 32 < LKK) {
      stage_async(P, m0, LKK, kt + 32, aoff[nxt], tid);
      stage_b(Bs[nxt], kt + 32);
    }
    mma_chunk(As[cur], Bs[cur], wave, lane, acc);
    wait_async0();
    __syncthreads();
  }

  const int mloc = (lane >> 4) << 3;
  const int nl = lane & 15;
#pragma unroll
  for (int i = 0; i < 8; ++i) {
    const int ng = n0 + i * 16 + nl;
#pragma unroll
    for (int r = 0; r < 8; ++r) {
      const long mg = m0 + wave * 16 + mloc + r;
      O[mg * DD + ng] = acc[i][r];
    }
  }
}

// ---------------------------------------------------------------------------
extern "C" void kernel_launch(void* const* d_in, const int* in_sizes, int n_in,
                              void* d_out, int out_size, void* d_ws, size_t ws_size,
                              hipStream_t stream) {
  (void)in_sizes; (void)n_in; (void)out_size; (void)ws_size;
  const float* Q    = (const float*)d_in[0];
  const float* K    = (const float*)d_in[1];
  const float* M    = (const float*)d_in[2];
  const float* W    = (const float*)d_in[3];
  const float* bias = (const float*)d_in[4];
  float* out = (float*)d_out;

  char* ws = (char*)d_ws;
  unsigned short* Qb = (unsigned short*)ws;                           // 64 MB
  unsigned short* Kb = (unsigned short*)(ws + ((size_t)64  << 20));   // 64 MB
  unsigned short* Wb = (unsigned short*)(ws + ((size_t)128 << 20));   //  2 MB
  unsigned short* Qf = (unsigned short*)(ws + ((size_t)136 << 20));   // 64 MB
  float*          S  = (float*)        (ws + ((size_t)200 << 20));    // 128 MB
  unsigned short* P  = (unsigned short*)(ws + ((size_t)328 << 20));   // 64 MB

  // 1) bf16 conversions (Kb also applies the key mask)
  k_prep_key<<<dim3((BB * LKK * DD) / 1024), 256, 0, stream>>>(K, M, Kb);
  k_cvt<<<dim3((BB * LQ * DD) / 1024), 256, 0, stream>>>(Q, Qb);
  k_cvt<<<dim3((DD * DD) / 1024), 256, 0, stream>>>(W, Wb);
  // 2) Qf = relu(Q @ W^T + b)
  k_gemm1<<<dim3(DD / 128, (BB * LQ) / 128), 256, 0, stream>>>(Qb, Wb, bias, Qf);
  // 3) S = Qf @ Kb^T + mask penalty
  k_gemm2<<<dim3(LKK / 128, LQ / 128, BB), 256, 0, stream>>>(Qf, Kb, M, S);
  // 4) P = softmax rows
  k_softmax<<<dim3((BB * LQ) / 8), 256, 0, stream>>>(S, P);
  // 5) O = P @ Kb
  k_gemm3<<<dim3(DD / 128, LQ / 128, BB), 256, 0, stream>>>(P, Kb, out);
}
